// BHCN_AWX_55190329754262
// MI455X (gfx1250) — matrix operand order repaired and checked by reading, not compiled, not run on hardware
//
#include <hip/hip_runtime.h>
#include <hip/hip_bf16.h>

// ---------------------------------------------------------------------------
// BHCN-AWX forward for MI455X (gfx1250, wave32, WMMA bf16)
//   l1 = x@W0^T+b0 ; out0 = log_softmax(l1)
//   cat1 = [LN(relu(l1)), x] ; l2 = cat1@W1^T+b1 ; out1 = log_softmax(l2)
//   cat2 = [LN(relu(l2)), x] ; l3 = cat2@W2^T+b2 ; out2 = log_softmax(l3)
//   awx = clip(sigmoid(l3)^2 @ R^T, eps, 1-eps)^(1/2)
// ---------------------------------------------------------------------------

typedef __attribute__((ext_vector_type(16))) __bf16       v16bf;
typedef __attribute__((ext_vector_type(8)))  float        v8f;
typedef __attribute__((ext_vector_type(8)))  unsigned int v8u;

#define BATCH   1024
#define XDIM    768
#define TOTAL_C 8736

__device__ __forceinline__ unsigned int pack2_bf16(float a, float b) {
  unsigned int ua = __builtin_bit_cast(unsigned int, a);
  unsigned int ub = __builtin_bit_cast(unsigned int, b);
  ua += 0x7FFFu + ((ua >> 16) & 1u);   // round-to-nearest-even
  ub += 0x7FFFu + ((ub >> 16) & 1u);
  return (ua >> 16) | (ub & 0xFFFF0000u);
}

// ---------------------------------------------------------------------------
// Tiled GEMM:  C[M,N] = A[M,K] * W[N,K]^T (+bias) with optional AWX epilogue.
// A,W are f32 in global; converted to bf16 while staging to LDS (b128 loads).
// K must be a multiple of KCH*32 and of 4; M a multiple of BM. N-tail guarded.
// ---------------------------------------------------------------------------
template <int BM, int BN, int WAVES_M, int WAVES_N, int WM_TILES, int WN_TILES,
          int KCH>
__global__ __launch_bounds__(WAVES_M * WAVES_N * 32)
void k_gemm_bf16(const float* __restrict__ A, const float* __restrict__ W,
                 const float* __restrict__ bias, float* __restrict__ C,
                 int M, int N, int K, int ldc, int epilogue) {
  constexpr int NT    = WAVES_M * WAVES_N * 32;
  constexpr int KSTEP = KCH * 32;            // K elements staged per iteration
  constexpr int LDW   = KCH * 16 + 2;        // u32 per LDS row (+2 pad)

  __shared__ unsigned int aLds[BM * LDW];
  __shared__ unsigned int bLds[BN * LDW];

  const int tid  = threadIdx.x;
  const int lane = tid & 31;
  const int wv   = tid >> 5;
  const int wm   = wv / WAVES_N;
  const int wn   = wv % WAVES_N;
  const int half = lane >> 4;
  const int l15  = lane & 15;

  const int m0 = blockIdx.y * BM;
  const int n0 = blockIdx.x * BN;

  v8f acc[WM_TILES][WN_TILES];
#pragma unroll
  for (int tm = 0; tm < WM_TILES; ++tm)
#pragma unroll
    for (int tn = 0; tn < WN_TILES; ++tn)
      acc[tm][tn] = (v8f){0.f, 0.f, 0.f, 0.f, 0.f, 0.f, 0.f, 0.f};

  for (int kb = 0; kb < K; kb += KSTEP) {
    // ---- stage A tile (BM x KSTEP f32 -> packed bf16 pairs), b128 loads ----
    for (int e = tid; e < BM * KCH * 8; e += NT) {
      const int r = e / (KCH * 8), q = e % (KCH * 8);  // q: float4 within row
      const float4 f =
          *(const float4*)(A + (size_t)(m0 + r) * K + kb + 4 * q);
      uint2 p;
      p.x = pack2_bf16(f.x, f.y);
      p.y = pack2_bf16(f.z, f.w);
      *(uint2*)(&aLds[r * LDW + 2 * q]) = p;
    }
    // ---- stage W tile (BN x KSTEP), N-tail zero-padded ----
    for (int e = tid; e < BN * KCH * 8; e += NT) {
      const int r = e / (KCH * 8), q = e % (KCH * 8);
      const int n = n0 + r;
      float4 f = {0.f, 0.f, 0.f, 0.f};
      if (n < N) f = *(const float4*)(W + (size_t)n * K + kb + 4 * q);
      uint2 p;
      p.x = pack2_bf16(f.x, f.y);
      p.y = pack2_bf16(f.z, f.w);
      *(uint2*)(&bLds[r * LDW + 2 * q]) = p;
    }
    // ---- prefetch next W K-tile (one 128B line per thread covers it) ----
    {
      const int kbn = kb + KSTEP;
      if (kbn < K) {
        for (int e = tid; e < BN * KCH; e += NT) {
          const int r = e / KCH, l = e % KCH;  // l: 128B line within row
          const int n = n0 + r;
          if (n < N)
            __builtin_prefetch(W + (size_t)n * K + kbn + 32 * l, 0, 0);
        }
      }
    }
    __syncthreads();

    // ---- per K-chunk: gather fragments (ISA VGPR layouts) and WMMA ----
#pragma unroll
    for (int c = 0; c < KCH; ++c) {
      const int cb = 16 * c;  // u32 offset of this 32-wide K chunk
      v8u af[WM_TILES];
#pragma unroll
      for (int tm = 0; tm < WM_TILES; ++tm) {
        const int m = wm * (WM_TILES * 16) + tm * 16 + l15;
#pragma unroll
        for (int i = 0; i < 8; ++i) {
          const int idx = cb + (i < 4 ? i : i + 4) + 4 * half;  // A 16x32 bf16
          af[tm][i] = aLds[m * LDW + idx];
        }
      }
      v8u bf[WN_TILES];
#pragma unroll
      for (int tn = 0; tn < WN_TILES; ++tn) {
        const int n = wn * (WN_TILES * 16) + tn * 16 + l15;
#pragma unroll
        for (int i = 0; i < 8; ++i) {
          const int idx = cb + (i < 4 ? i : i + 4) + 4 * half;  // B 32x16 bf16
          bf[tn][i] = bLds[n * LDW + idx];
        }
      }
#pragma unroll
      for (int tm = 0; tm < WM_TILES; ++tm)
#pragma unroll
        for (int tn = 0; tn < WN_TILES; ++tn)
          acc[tm][tn] = __builtin_amdgcn_wmma_f32_16x16x32_bf16(
              false, __builtin_bit_cast(v16bf, af[tm]),
              false, __builtin_bit_cast(v16bf, bf[tn]),
              (short)0, acc[tm][tn], false, false);
    }
    __syncthreads();
  }

  // ---- epilogue: C/D layout m = i + 8*half, n = lane&15 ----
#pragma unroll
  for (int tm = 0; tm < WM_TILES; ++tm) {
#pragma unroll
    for (int tn = 0; tn < WN_TILES; ++tn) {
      const int col = n0 + wn * (WN_TILES * 16) + tn * 16 + l15;
      if (col < N) {
        const float bv = bias ? bias[col] : 0.f;
#pragma unroll
        for (int i = 0; i < 8; ++i) {
          const int row = m0 + wm * (WM_TILES * 16) + tm * 16 + i + 8 * half;
          float v = acc[tm][tn][i] + bv;
          if (epilogue == 1) {  // AWX: clip then ^(1/2)
            v = fminf(fmaxf(v, 1e-6f), 1.f - 1e-6f);
            v = sqrtf(v);
          }
          C[(size_t)row * ldc + col] = v;
        }
      }
    }
  }
}

// ---------------------------------------------------------------------------
// Row-wise log_softmax: one row per block; out already offset to column start.
// ---------------------------------------------------------------------------
__global__ __launch_bounds__(256)
void k_logsoftmax(const float* __restrict__ in, float* __restrict__ out,
                  int N, int ldo) {
  __shared__ float red[256];
  const int row = blockIdx.x, tid = threadIdx.x;
  const float* r = in + (size_t)row * N;

  float m = -INFINITY;
  for (int c = tid; c < N; c += 256) m = fmaxf(m, r[c]);
  red[tid] = m;
  __syncthreads();
  for (int s = 128; s > 0; s >>= 1) {
    if (tid < s) red[tid] = fmaxf(red[tid], red[tid + s]);
    __syncthreads();
  }
  m = red[0];
  __syncthreads();

  float sum = 0.f;
  for (int c = tid; c < N; c += 256) sum += expf(r[c] - m);
  red[tid] = sum;
  __syncthreads();
  for (int s = 128; s > 0; s >>= 1) {
    if (tid < s) red[tid] += red[tid + s];
    __syncthreads();
  }
  const float lse = m + logf(red[0]);

  for (int c = tid; c < N; c += 256) out[(size_t)row * ldo + c] = r[c] - lse;
}

// ---------------------------------------------------------------------------
// cat = [ LayerNorm(relu(l), no-affine), x ] — one row per block.
// ---------------------------------------------------------------------------
__global__ __launch_bounds__(256)
void k_ln_concat(const float* __restrict__ l, const float* __restrict__ x,
                 float* __restrict__ cat, int D) {
  __shared__ float rs[256];
  __shared__ float rq[256];
  const int row = blockIdx.x, tid = threadIdx.x;
  const float* lr = l + (size_t)row * D;

  float s = 0.f, q = 0.f;
  for (int c = tid; c < D; c += 256) {
    const float h = fmaxf(lr[c], 0.f);
    s += h; q += h * h;
  }
  rs[tid] = s; rq[tid] = q;
  __syncthreads();
  for (int st = 128; st > 0; st >>= 1) {
    if (tid < st) { rs[tid] += rs[tid + st]; rq[tid] += rq[tid + st]; }
    __syncthreads();
  }
  const float mu  = rs[0] / (float)D;
  const float var = rq[0] / (float)D - mu * mu;   // biased var, as jnp.var
  const float inv = rsqrtf(var + 1e-5f);

  float* cr = cat + (size_t)row * (D + XDIM);
  for (int c = tid; c < D; c += 256) {
    const float h = fmaxf(lr[c], 0.f);
    cr[c] = (h - mu) * inv;
  }
  const float* xr = x + (size_t)row * XDIM;
  for (int c = tid; c < XDIM; c += 256) cr[D + c] = xr[c];
}

// ---------------------------------------------------------------------------
// s2 = sigmoid(v)^2 elementwise
// ---------------------------------------------------------------------------
__global__ __launch_bounds__(256)
void k_sig2(const float* __restrict__ in, float* __restrict__ out, int n) {
  const int i = blockIdx.x * 256 + threadIdx.x;
  if (i < n) {
    const float s = 1.f / (1.f + expf(-in[i]));
    out[i] = s * s;
  }
}

// ---------------------------------------------------------------------------
extern "C" void kernel_launch(void* const* d_in, const int* in_sizes, int n_in,
                              void* d_out, int out_size, void* d_ws,
                              size_t ws_size, hipStream_t stream) {
  (void)in_sizes; (void)n_in; (void)out_size; (void)ws_size;

  const float* x  = (const float*)d_in[0];
  const float* W0 = (const float*)d_in[1];
  const float* b0 = (const float*)d_in[2];
  const float* W1 = (const float*)d_in[3];
  const float* b1 = (const float*)d_in[4];
  const float* W2 = (const float*)d_in[5];
  const float* b2 = (const float*)d_in[6];
  const float* R  = (const float*)d_in[7];

  float* out = (float*)d_out;
  float* ws  = (float*)d_ws;

  // workspace layout (f32 elements)
  float* l1   = ws;                               // 1024*32
  float* cat1 = l1   + (size_t)BATCH * 32;        // 1024*800
  float* l2   = cat1 + (size_t)BATCH * 800;       // 1024*512
  float* cat2 = l2   + (size_t)BATCH * 512;       // 1024*1280
  float* l3   = cat2 + (size_t)BATCH * 1280;      // 1024*8192
  float* s2   = l3   + (size_t)BATCH * 8192;      // 1024*8192

  float* local = out;                             // (1024, 8736)
  float* awx   = out + (size_t)BATCH * TOTAL_C;   // (1024, 8736)

  // Level 1: l1 = x @ W0^T + b0   [M=1024, N=32, K=768]  (768 % 64 == 0)
  {
    dim3 g(1, BATCH / 128);
    k_gemm_bf16<128, 32, 4, 1, 2, 2, 2><<<g, 128, 0, stream>>>(
        x, W0, b0, l1, BATCH, 32, XDIM, 32, 0);
  }
  k_logsoftmax<<<BATCH, 256, 0, stream>>>(l1, local + 0, 32, TOTAL_C);
  k_ln_concat <<<BATCH, 256, 0, stream>>>(l1, x, cat1, 32);

  // Level 2: l2 = cat1 @ W1^T + b1   [N=512, K=800]  (800 % 64 != 0 -> KCH=1)
  {
    dim3 g(512 / 128, BATCH / 128);
    k_gemm_bf16<128, 128, 2, 4, 4, 2, 1><<<g, 256, 0, stream>>>(
        cat1, W1, b1, l2, BATCH, 512, 800, 512, 0);
  }
  k_logsoftmax<<<BATCH, 256, 0, stream>>>(l2, local + 32, 512, TOTAL_C);
  k_ln_concat <<<BATCH, 256, 0, stream>>>(l2, x, cat2, 512);

  // Level 3: l3 = cat2 @ W2^T + b2   [N=8192, K=1280]  (1280 % 64 == 0)
  {
    dim3 g(8192 / 128, BATCH / 128);
    k_gemm_bf16<128, 128, 2, 4, 4, 2, 2><<<g, 256, 0, stream>>>(
        cat2, W2, b2, l3, BATCH, 8192, 1280, 8192, 0);
  }
  k_logsoftmax<<<BATCH, 256, 0, stream>>>(l3, local + 544, 8192, TOTAL_C);

  // AWX: awx = clip(sigmoid(l3)^2 @ R^T)^(1/2)   [N=8736, K=8192]
  k_sig2<<<(BATCH * 8192 + 255) / 256, 256, 0, stream>>>(l3, s2, BATCH * 8192);
  {
    dim3 g((TOTAL_C + 127) / 128, BATCH / 128);
    k_gemm_bf16<128, 128, 2, 4, 4, 2, 2><<<g, 256, 0, stream>>>(
        s2, R, nullptr, awx, BATCH, TOTAL_C, 8192, TOTAL_C, 1);
  }
}
